// RNN_63024350101516
// MI455X (gfx1250) — matrix operand order repaired and checked by reading, not compiled, not run on hardware
//
#include <hip/hip_runtime.h>
#include <math.h>

// ---------------------------------------------------------------------------
// RNN LM for MI455X (gfx1250, wave32).
//   B=8, S=512, K=512, H=1024, V=32000
// Strategy:
//   - ux GEMM + recurrence GEMM: exact V_WMMA_F32_16X16X4_F32
//   - vocab GEMM (268 GFLOP, dominant): V_WMMA_F32_16X16X32_BF16 (bf16 in,
//     f32 accumulate). Output write (524 MB) is the HBM floor (~24us @
//     23.3 TB/s); V_w (64 MB bf16) stays resident in the 192 MB L2.
//     B panels are staged global->LDS with GLOBAL_LOAD_ASYNC_TO_LDS_B128
//     (ASYNCcnt) and shared by 8 waves/block => 8x less L2 read traffic,
//     double-buffered to overlap the async DMA with WMMA issue.
//   - recurrence: persistent 16-block kernel + software grid barrier
//     (1 barrier per step, no 512 kernel launches).
// Workspace: ~91 MB (ux f32 16MB, h dbl-buf 128KB, hs bf16 8MB, Vw bf16 64MB).
// ---------------------------------------------------------------------------

typedef __attribute__((ext_vector_type(2)))  float  v2f;
typedef __attribute__((ext_vector_type(8)))  float  v8f;
typedef __attribute__((ext_vector_type(8)))  __bf16 v8bf;
typedef __attribute__((ext_vector_type(16))) __bf16 v16bf;

#define RB 8
#define RS 512
#define RK 512
#define RH 1024
#define RV 32000
#define NBLK_RNN 16u

// ws byte offsets (all 256B aligned)
#define UX_OFF   0LL                                   // f32 [S][B][H]  16 MB
#define HB0_OFF  (UX_OFF  + (long long)RS*RB*RH*4)     // f32 [16][H]
#define HB1_OFF  (HB0_OFF + 16LL*RH*4)                 // f32 [16][H]
#define HSB_OFF  (HB1_OFF + 16LL*RH*4)                 // bf16 [B*S][H]   8 MB
#define VWB_OFF  (HSB_OFF + (long long)RB*RS*RH*2)     // bf16 [V][H]    64 MB
#define BAR_OFF  (VWB_OFF + (long long)RV*RH*2)        // 2 x u32

__device__ inline unsigned short f32_to_bf16_rne(float f) {
  unsigned u = __float_as_uint(f);
  u += 0x7FFFu + ((u >> 16) & 1u);
  return (unsigned short)(u >> 16);
}

// ---------------------------------------------------------------------------
__global__ void k_init(const float* __restrict__ h0, float* __restrict__ hb0,
                       float* __restrict__ hb1, unsigned* __restrict__ bar) {
  int idx = blockIdx.x * blockDim.x + threadIdx.x;
  if (idx < 16 * RH) {
    int row = idx / RH, col = idx % RH;
    hb0[idx] = (row < RB) ? h0[row * RH + col] : 0.0f;  // pad rows 8..15 = 0
    hb1[idx] = 0.0f;
  }
  if (idx == 0) { bar[0] = 0u; bar[1] = 0u; }
}

__global__ void k_cast_vw(const float* __restrict__ Vw,
                          unsigned short* __restrict__ Vwb) {
  long long n = (long long)RV * RH;
  long long stride = (long long)gridDim.x * blockDim.x;
  for (long long i = (long long)blockIdx.x * blockDim.x + threadIdx.x; i < n; i += stride)
    Vwb[i] = f32_to_bf16_rne(Vw[i]);
}

// ---------------------------------------------------------------------------
// ux[s][b][h] = sum_k x[b][s][k] * U_w[h][k] + U_b[h]
// f32 WMMA 16x16x4, one 16x16 output tile per wave.
__global__ void __launch_bounds__(256) k_ux(const float* __restrict__ x,
                                            const float* __restrict__ Uw,
                                            const float* __restrict__ Ub,
                                            float* __restrict__ ux) {
  const int lane = threadIdx.x & 31;
  const int tid  = blockIdx.x * 8 + (threadIdx.x >> 5);   // tile id, 16384 total
  const int mt = tid >> 6;                 // 0..255 (rows of [4096])
  const int nt = tid & 63;                 // 0..63  (cols of [1024])
  const int r0 = mt * 16;
  const int b  = r0 >> 9;                  // r = b*512 + s
  const int s0 = r0 & 511;
  const int h0t = nt * 16;
  const int l15 = lane & 15;
  const int koff = (lane >= 16) ? 2 : 0;

  const float* ap = x  + (long long)(r0 + l15) * RK + koff;
  const float* bp = Uw + (long long)(h0t + l15) * RK + koff;

  v8f c = {};
#pragma unroll 4
  for (int k = 0; k < RK; k += 4) {
    v2f a  = *(const v2f*)(ap + k);
    v2f bb = *(const v2f*)(bp + k);
    c = __builtin_amdgcn_wmma_f32_16x16x4_f32(false, a, false, bb,
                                              (short)0, c, false, false);
  }
  const int h = h0t + l15;
  const float bias = Ub[h];
  const int mhi = (lane >= 16) ? 8 : 0;    // C: lanes16-31 hold M = 8+r
#pragma unroll
  for (int r = 0; r < 8; ++r) {
    int s = s0 + r + mhi;
    ux[(long long)(s * RB + b) * RH + h] = c[r] + bias;
  }
}

// ---------------------------------------------------------------------------
__device__ inline void grid_barrier(unsigned* arrive, unsigned* gen) {
  __syncthreads();
  if (threadIdx.x == 0) {
    __threadfence();
    volatile unsigned* vgen = gen;
    unsigned g = *vgen;
    if (atomicAdd(arrive, 1u) == NBLK_RNN - 1u) {
      *arrive = 0u;
      __threadfence();
      atomicAdd(gen, 1u);
    } else {
      while (*vgen == g) __builtin_amdgcn_s_sleep(2);
    }
    __threadfence();
  }
  __syncthreads();
}

// h_{s+1} = tanh(ux_s + h_s @ W^T + W_b), M padded 8->16 (pad rows stay 0).
// 16 blocks x 128 threads = 64 waves; wave gw owns output columns [gw*16,+16).
__global__ void __launch_bounds__(128) k_rnn(const float* __restrict__ Ww,
                                             const float* __restrict__ Wb,
                                             const float* __restrict__ ux,
                                             float* __restrict__ hb0,
                                             float* __restrict__ hb1,
                                             unsigned short* __restrict__ hsb,
                                             float* __restrict__ hlast,
                                             unsigned* __restrict__ bar) {
  const int lane = threadIdx.x & 31;
  const int gw   = blockIdx.x * 4 + (threadIdx.x >> 5);   // 0..63
  const int j0   = gw * 16;
  const int l15  = lane & 15;
  const int koff = (lane >= 16) ? 2 : 0;
  const int j    = j0 + l15;
  const float bias = Wb[j];
  const float* wp = Ww + (long long)j * RH + koff;        // B[K=i][N=j]=W[j][i]

  float* bufs[2] = { hb0, hb1 };
  for (int s = 0; s < RS; ++s) {
    const float* hp = bufs[s & 1];
    float*       hn = bufs[(s + 1) & 1];
    const float* apx = hp + l15 * RH + koff;
    v8f c = {};
#pragma unroll 8
    for (int k = 0; k < RH; k += 4) {
      v2f a = *(const v2f*)(apx + k);
      v2f w = *(const v2f*)(wp + k);
      c = __builtin_amdgcn_wmma_f32_16x16x4_f32(false, a, false, w,
                                                (short)0, c, false, false);
    }
    if (lane < 16) {                       // lanes16-31 carry pad rows M=8..15
#pragma unroll
      for (int r = 0; r < RB; ++r) {       // M = r = batch index
        float v = tanhf(c[r] + bias + ux[(long long)(s * RB + r) * RH + j]);
        hn[r * RH + j] = v;
        hsb[(long long)(r * RS + s) * RH + j] = f32_to_bf16_rne(v);
      }
    }
    grid_barrier(bar, bar + 1);
  }
  if (blockIdx.x == 0) {                   // final h lives in bufs[S&1]=bufs[0]
    const float* hf = bufs[RS & 1];
    for (int i = threadIdx.x; i < RB * RH; i += blockDim.x) hlast[i] = hf[i];
  }
}

// ---------------------------------------------------------------------------
// logits = hs @ V_w^T + V_b via bf16 WMMA 16x16x32 (f32 acc).
// Block = 8 waves = 128 rows x 64 cols macro-tile. The 64x32 bf16 B panel
// (4 KB) is staged global->LDS with GLOBAL_LOAD_ASYNC_TO_LDS_B128 (one b128
// per thread, ASYNCcnt tracked), double-buffered, and shared by all 8 waves
// => 8x less L2 read traffic for V_w. A fragments stream from global (L2-hot).
// Fragment layouts:
//   A (bf16 16x32): lane=row M=lane&15; halves[0..15]=K+(lane>=16?16:0)
//   B (bf16 32x16): lane=col N=lane&15; halves[0..15]=K+(lane>=16?16:0)
__global__ void __launch_bounds__(256) k_logits(const unsigned short* __restrict__ hsb,
                                                const unsigned short* __restrict__ Vwb,
                                                const float* __restrict__ Vb,
                                                float* __restrict__ out) {
  __shared__ unsigned short sB[2][64 * 32];   // [buf][row(64)][k(32)] bf16, 4KB each

  const int lane = threadIdx.x & 31;
  const int wave = threadIdx.x >> 5;
  const int mg = blockIdx.x / 500;            // 0..31  (m-groups of 8 tiles)
  const int ng = blockIdx.x - mg * 500;       // 0..499 (64-col groups)
  const int m0 = (mg * 8 + wave) * 16;        // this wave's 16 rows
  const int n0 = ng * 64;
  const int l15  = lane & 15;
  const int ka   = (lane >= 16) ? 8  : 0;
  const int kb   = (lane >= 16) ? 16 : 0;

  // Per-thread staging slot: 64 rows x 64B/row = 256 threads x 16B.
  const int srow  = threadIdx.x >> 2;          // 0..63
  const int spart = (threadIdx.x & 3) * 8;     // half-offset within row chunk
  const unsigned short* sbase = Vwb + (long long)(n0 + srow) * RH + spart;

  const unsigned short* ap = hsb + (long long)(m0 + l15) * RH;

  // async-stage the 64x32 B panel for k-chunk into LDS buffer `buf`
#define STAGE_B(bufi, kk)                                                     \
  {                                                                           \
    unsigned ldso = (unsigned)(size_t)(&sB[(bufi)][srow * 32 + spart]);       \
    const unsigned short* gsrc = sbase + (kk);                                \
    asm volatile("global_load_async_to_lds_b128 %0, %1, off"                  \
                 :: "v"(ldso), "v"(gsrc) : "memory");                         \
  }

  STAGE_B(0, 0)
  asm volatile("s_wait_asynccnt 0" ::: "memory");
  __syncthreads();

  v8f c0 = {}, c1 = {}, c2 = {}, c3 = {};
  int buf = 0;
  for (int k = 0; k < RH; k += 32) {
    if (k + 32 < RH) STAGE_B(buf ^ 1, k + 32)    // overlap DMA with WMMA

    v8bf alo = *(const v8bf*)(ap + k + kb);
    v8bf ahi = *(const v8bf*)(ap + k + kb + 8);
    v16bf a = __builtin_shufflevector(alo, ahi, 0, 1, 2, 3, 4, 5, 6, 7,
                                                8, 9, 10, 11, 12, 13, 14, 15);
    const unsigned short* bb = &sB[buf][0];
    v16bf b0 = *(const v16bf*)(bb + (0 * 16 + l15) * 32 + kb);
    c0 = __builtin_amdgcn_wmma_f32_16x16x32_bf16(false, a, false, b0, (short)0, c0, false, false);
    v16bf b1 = *(const v16bf*)(bb + (1 * 16 + l15) * 32 + kb);
    c1 = __builtin_amdgcn_wmma_f32_16x16x32_bf16(false, a, false, b1, (short)0, c1, false, false);
    v16bf b2 = *(const v16bf*)(bb + (2 * 16 + l15) * 32 + kb);
    c2 = __builtin_amdgcn_wmma_f32_16x16x32_bf16(false, a, false, b2, (short)0, c2, false, false);
    v16bf b3 = *(const v16bf*)(bb + (3 * 16 + l15) * 32 + kb);
    c3 = __builtin_amdgcn_wmma_f32_16x16x32_bf16(false, a, false, b3, (short)0, c3, false, false);

    asm volatile("s_wait_asynccnt 0" ::: "memory");  // next panel landed
    __syncthreads();                                 // all waves done with buf
    buf ^= 1;
  }
#undef STAGE_B

  const int mhi = (lane >= 16) ? 8 : 0;
#define STORE_TILE(cx, t)                                                     \
  {                                                                           \
    int col = n0 + (t) * 16 + l15;                                            \
    float bias = Vb[col];                                                     \
    _Pragma("unroll") for (int r = 0; r < 8; ++r)                             \
        out[(long long)(m0 + r + mhi) * RV + col] = (cx)[r] + bias;           \
  }
  STORE_TILE(c0, 0) STORE_TILE(c1, 1) STORE_TILE(c2, 2) STORE_TILE(c3, 3)
#undef STORE_TILE
}

// ---------------------------------------------------------------------------
__global__ void __launch_bounds__(256) k_logsoftmax(float* __restrict__ out) {
  __shared__ float red[256];
  float* p = out + (long long)blockIdx.x * RV;
  float m = -INFINITY;
  for (int i = threadIdx.x; i < RV; i += 256) m = fmaxf(m, p[i]);
  red[threadIdx.x] = m;
  __syncthreads();
  for (int st = 128; st > 0; st >>= 1) {
    if (threadIdx.x < st) red[threadIdx.x] = fmaxf(red[threadIdx.x], red[threadIdx.x + st]);
    __syncthreads();
  }
  m = red[0];
  __syncthreads();
  float s = 0.0f;
  for (int i = threadIdx.x; i < RV; i += 256) s += expf(p[i] - m);
  red[threadIdx.x] = s;
  __syncthreads();
  for (int st = 128; st > 0; st >>= 1) {
    if (threadIdx.x < st) red[threadIdx.x] += red[threadIdx.x + st];
    __syncthreads();
  }
  const float lse = m + logf(red[0]);
  for (int i = threadIdx.x; i < RV; i += 256) p[i] -= lse;
}

// ---------------------------------------------------------------------------
extern "C" void kernel_launch(void* const* d_in, const int* in_sizes, int n_in,
                              void* d_out, int out_size, void* d_ws, size_t ws_size,
                              hipStream_t stream) {
  const float* x  = (const float*)d_in[0];
  const float* h0 = (const float*)d_in[1];
  const float* Uw = (const float*)d_in[2];
  const float* Ub = (const float*)d_in[3];
  const float* Ww = (const float*)d_in[4];
  const float* Wb = (const float*)d_in[5];
  const float* Vw = (const float*)d_in[6];
  const float* Vb = (const float*)d_in[7];
  (void)in_sizes; (void)n_in; (void)out_size; (void)ws_size;

  float* out = (float*)d_out;                              // [B*S, V] then [B,H]
  char* ws = (char*)d_ws;                                  // needs ~91 MB
  float*          ux   = (float*)(ws + UX_OFF);
  float*          hb0  = (float*)(ws + HB0_OFF);
  float*          hb1  = (float*)(ws + HB1_OFF);
  unsigned short* hsb  = (unsigned short*)(ws + HSB_OFF);
  unsigned short* vwb  = (unsigned short*)(ws + VWB_OFF);
  unsigned*       bar  = (unsigned*)(ws + BAR_OFF);
  float*          hlast = out + (long long)RB * RS * RV;

  k_init      <<<64,    256, 0, stream>>>(h0, hb0, hb1, bar);
  k_cast_vw   <<<8192,  256, 0, stream>>>(Vw, vwb);
  k_ux        <<<2048,  256, 0, stream>>>(x, Uw, Ub, ux);
  k_rnn       <<<NBLK_RNN, 128, 0, stream>>>(Ww, Wb, ux, hb0, hb1, hsb, hlast, bar);
  k_logits    <<<16000, 256, 0, stream>>>(hsb, vwb, Vb, out);
  k_logsoftmax<<<RB * RS, 256, 0, stream>>>(out);
}
